// SingleAttention_38208029065631
// MI455X (gfx1250) — matrix operand order repaired and checked by reading, not compiled, not run on hardware
//
#include <hip/hip_runtime.h>
#include <hip/hip_bf16.h>
#include <stdint.h>

// ---------------------------------------------------------------------------
// SingleAttention for MI455X (gfx1250): flash-style, bf16 WMMA / f32 accum,
// TDM (tensor_load_to_lds) double-buffered K/V staging shared by 4 waves.
//   B=8, T=2048, D=512, DK=DV=128
// ---------------------------------------------------------------------------

#define T_DIM 2048
#define D_DIM 512
#define DK_DIM 128
#define B_DIM 8
#define NSB (T_DIM / 32)   // 64 key-steps of 32 keys

typedef __attribute__((ext_vector_type(16))) __bf16       v16bf;
typedef __attribute__((ext_vector_type(8)))  float        v8f;
typedef __attribute__((ext_vector_type(4)))  unsigned int u32x4;

union BF16x16 {
    v16bf          v;
    u32x4          q[2];
    unsigned short s[16];
    __device__ BF16x16() {}
};

__device__ __forceinline__ unsigned short f2bf_rne(float f) {
    unsigned int u = __float_as_uint(f);
    u += 0x7FFFu + ((u >> 16) & 1u);
    return (unsigned short)(u >> 16);
}

#if __has_builtin(__builtin_amdgcn_tensor_load_to_lds)
#define USE_TDM 1
#else
#define USE_TDM 0
#endif

// ---------------------------------------------------------------------------
// TDM 2-D tile load: builds a D# (CDNA5 ISA) for a
// 16-bit-element 2-D tensor and issues TENSOR_LOAD_TO_LDS. Completion is
// tracked with TENSORcnt. Caller must be a full uniform wave.
// ---------------------------------------------------------------------------
#if USE_TDM
__device__ __forceinline__ void tdm_load_2d(unsigned lds_off, const void* gptr,
                                            unsigned tensor_d0, unsigned tensor_d1,
                                            unsigned tile_d0, unsigned tile_d1,
                                            unsigned long long stride0_elems) {
    typedef __attribute__((ext_vector_type(4))) unsigned int tdm_u4;
    typedef __attribute__((ext_vector_type(8))) int          tdm_i8;
    typedef __attribute__((ext_vector_type(4))) int          tdm_i4;

    unsigned long long ga = (unsigned long long)gptr;
    tdm_u4 g0;
    g0.x = 1u;                                                   // count=1 (valid), user mode
    g0.y = lds_off;                                              // LDS byte address
    g0.z = (unsigned)(ga & 0xFFFFFFFFu);                         // global_addr[31:0]
    g0.w = ((unsigned)(ga >> 32) & 0x01FFFFFFu) | 0x80000000u;   // addr[56:32] | type=2

    tdm_i8 g1;
    g1[0] = (int)(1u << 16);                                     // data_size=1 (2 bytes)
    g1[1] = (int)((tensor_d0 & 0xFFFFu) << 16);                  // tensor_dim0[15:0]
    g1[2] = (int)(((tensor_d0 >> 16) & 0xFFFFu) |
                  ((tensor_d1 & 0xFFFFu) << 16));                // dim0[31:16] | dim1[15:0]
    g1[3] = (int)(((tensor_d1 >> 16) & 0xFFFFu) |
                  ((tile_d0 & 0xFFFFu) << 16));                  // dim1[31:16] | tile_dim0
    g1[4] = (int)(tile_d1 & 0xFFFFu);                            // tile_dim1 | tile_dim2=0
    g1[5] = (int)(unsigned)(stride0_elems & 0xFFFFFFFFull);      // tensor_dim0_stride[31:0]
    g1[6] = (int)(unsigned)((stride0_elems >> 32) & 0xFFFFull);  // stride0[47:32] | stride1=0
    g1[7] = 0;

    tdm_i4 gz = {0, 0, 0, 0};                                    // groups 2/3: <=2-D tensor
#if defined(__clang_major__) && __clang_major__ >= 23
    tdm_i8 gz8 = {0, 0, 0, 0, 0, 0, 0, 0};
    __builtin_amdgcn_tensor_load_to_lds(g0, g1, gz, gz, gz8, 0);
#else
    __builtin_amdgcn_tensor_load_to_lds(g0, g1, gz, gz, 0);
#endif
}
#endif

__device__ __forceinline__ unsigned lds_offset_of(const void* p) {
    // Generic LDS pointers carry the wave-relative LDS byte offset in the
    // low 32 bits (aperture in the high bits) — ISA §10.2.
    return (unsigned)(unsigned long long)p;
}

// ---------------------------------------------------------------------------
// Kernel 0: transpose+convert weights W[512][128] f32 -> WT[128][512] bf16
// ---------------------------------------------------------------------------
__global__ void wt_kernel(const float* __restrict__ Wq, const float* __restrict__ Wk,
                          const float* __restrict__ Wv,
                          unsigned short* __restrict__ WTq, unsigned short* __restrict__ WTk,
                          unsigned short* __restrict__ WTv) {
    const float* W;
    unsigned short* WT;
    if (blockIdx.y == 0)      { W = Wq; WT = WTq; }
    else if (blockIdx.y == 1) { W = Wk; WT = WTk; }
    else                      { W = Wv; WT = WTv; }
    int idx = blockIdx.x * blockDim.x + threadIdx.x;
    int d = idx & (D_DIM - 1);
    int n = idx >> 9;
    WT[n * D_DIM + d] = f2bf_rne(W[d * DK_DIM + n]);
}

// ---------------------------------------------------------------------------
// Kernel 1: projections. One wave32 per 16x128 output tile.
//   z=0: q -> qbf [b][t][128]  z=1: k -> kbf [b][t][128]  z=2: v -> vT [b][dv][t]
// ---------------------------------------------------------------------------
__global__ void proj_kernel(const float* __restrict__ inq, const float* __restrict__ ink,
                            const float* __restrict__ inv,
                            const unsigned short* __restrict__ WTq,
                            const unsigned short* __restrict__ WTk,
                            const unsigned short* __restrict__ WTv,
                            const float* __restrict__ bq, const float* __restrict__ bk,
                            const float* __restrict__ bv,
                            unsigned short* __restrict__ qbf, unsigned short* __restrict__ kbf,
                            unsigned short* __restrict__ vT) {
    __shared__ unsigned short stage[16 * 128];

    const int tb   = blockIdx.x;
    const int b    = blockIdx.y;
    const int z    = blockIdx.z;
    const int lane = threadIdx.x;
    const int m    = lane & 15;
    const int h    = lane >> 4;

    const float* X; const unsigned short* WT; const float* bias;
    if (z == 0)      { X = inq; WT = WTq; bias = bq; }
    else if (z == 1) { X = ink; WT = WTk; bias = bk; }
    else             { X = inv; WT = WTv; bias = bv; }

    const v8f vzero = {0, 0, 0, 0, 0, 0, 0, 0};
    v8f acc[8];
#pragma unroll
    for (int nt = 0; nt < 8; ++nt) acc[nt] = vzero;

    const float* xrow = X + ((size_t)b * T_DIM + tb * 16 + m) * D_DIM;

    for (int kc = 0; kc < 16; ++kc) {
        BF16x16 af;
        const float* p0 = xrow + kc * 32 + 8 * h;
#pragma unroll
        for (int i = 0; i < 8; ++i) af.s[i] = f2bf_rne(p0[i]);
#pragma unroll
        for (int i = 0; i < 8; ++i) af.s[8 + i] = f2bf_rne(p0[16 + i]);

#pragma unroll
        for (int nt = 0; nt < 8; ++nt) {
            const unsigned short* wrow = WT + ((size_t)(nt * 16 + m)) * D_DIM + kc * 32 + 8 * h;
            BF16x16 bf;
            bf.q[0] = *(const u32x4*)(wrow);
            bf.q[1] = *(const u32x4*)(wrow + 16);
            acc[nt] = __builtin_amdgcn_wmma_f32_16x16x32_bf16(
                false, af.v, false, bf.v, (short)0, acc[nt], false, false);
        }
    }

    if (z < 2) {
#pragma unroll
        for (int nt = 0; nt < 8; ++nt) {
            float bb = bias[nt * 16 + m];
#pragma unroll
            for (int r = 0; r < 8; ++r)
                stage[(r + 8 * h) * 128 + nt * 16 + m] = f2bf_rne(acc[nt][r] + bb);
        }
        __syncthreads();
        unsigned short* base = (z == 0 ? qbf : kbf);
        unsigned short* dst  = base + ((size_t)b * T_DIM + tb * 16 + m) * DK_DIM + h * 64;
        const u32x4* src = (const u32x4*)(stage + m * 128 + h * 64);
        u32x4* gd = (u32x4*)dst;
#pragma unroll
        for (int i = 0; i < 8; ++i) gd[i] = src[i];
    } else {
#pragma unroll
        for (int nt = 0; nt < 8; ++nt) {
            float bb = bias[nt * 16 + m];
            BF16x16 pk;
#pragma unroll
            for (int r = 0; r < 8; ++r) pk.s[r] = f2bf_rne(acc[nt][r] + bb);
            u32x4* gd = (u32x4*)(vT + ((size_t)b * DK_DIM + nt * 16 + m) * T_DIM + tb * 16 + 8 * h);
            gd[0] = pk.q[0];
        }
    }
}

// ---------------------------------------------------------------------------
// Kernel 2: flash attention. 4 waves / block; each wave owns a 16-query tile.
// K/V tiles for each 32-key step are TDM-loaded into LDS (double-buffered)
// once per block and consumed by all 4 waves via ds_read_b128.
// ---------------------------------------------------------------------------
__global__ void __launch_bounds__(128)
attn_kernel(const unsigned short* __restrict__ qbf,
            const unsigned short* __restrict__ kbf,
            const unsigned short* __restrict__ vT,
            const int* __restrict__ maskp,
            float* __restrict__ out) {
    __shared__ unsigned short ktile[2][32 * 128];   // [buf][key][dk]    8KB each
    __shared__ unsigned short vtile[2][128 * 32];   // [buf][dv][key]    8KB each
    __shared__ unsigned short plds[4][16 * 32];     // per-wave P tile   1KB each

    const int wid  = threadIdx.x >> 5;              // wave id 0..3
    const int lane = threadIdx.x & 31;
    const int m    = lane & 15;
    const int h    = lane >> 4;
    const int tb   = blockIdx.x * 4 + wid;          // 16-query tile index
    const int b    = blockIdx.y;
    const int do_mask = *maskp;
    const float inv_sqrt = 0.08838834764831845f;    // 1/sqrt(128)

    const unsigned short* kbase = kbf + (size_t)b * T_DIM * DK_DIM;
    const unsigned short* vbase = vT + (size_t)b * DK_DIM * T_DIM;

#if USE_TDM
    // prologue: wave 0 issues TDM loads for key-step 0 into buffer 0
    if (threadIdx.x < 32) {
        tdm_load_2d(lds_offset_of(&ktile[0][0]), kbase,
                    DK_DIM, T_DIM, DK_DIM, 32, DK_DIM);
        tdm_load_2d(lds_offset_of(&vtile[0][0]), vbase,
                    T_DIM, DK_DIM, 32, DK_DIM, T_DIM);
    }
#endif

    // Q fragments for this wave's 16-row block
    v16bf qfr[4];
    const unsigned short* qrow = qbf + ((size_t)b * T_DIM + tb * 16 + m) * DK_DIM;
#pragma unroll
    for (int kc = 0; kc < 4; ++kc) {
        BF16x16 t;
        t.q[0] = *(const u32x4*)(qrow + kc * 32 + 8 * h);
        t.q[1] = *(const u32x4*)(qrow + kc * 32 + 16 + 8 * h);
        qfr[kc] = t.v;
    }

    const v8f vzero = {0, 0, 0, 0, 0, 0, 0, 0};
    v8f o[8];
#pragma unroll
    for (int nt = 0; nt < 8; ++nt) o[nt] = vzero;
    float mx[8], ls[8];
#pragma unroll
    for (int r = 0; r < 8; ++r) { mx[r] = -3.0e38f; ls[r] = 0.0f; }

    for (int sb = 0; sb < NSB; ++sb) {
        const int sbase = sb * 32;
        const int buf   = sb & 1;

#if USE_TDM
        __builtin_amdgcn_s_wait_tensorcnt(0);    // wave0's loads for tile sb done
        __syncthreads();                         // tile sb visible to all waves
        if (threadIdx.x < 32 && sb + 1 < NSB) {  // issue tile sb+1 into other buffer
            const int nb = buf ^ 1;
            tdm_load_2d(lds_offset_of(&ktile[nb][0]),
                        kbase + (size_t)(sbase + 32) * DK_DIM,
                        DK_DIM, T_DIM, DK_DIM, 32, DK_DIM);
            tdm_load_2d(lds_offset_of(&vtile[nb][0]),
                        vbase + (sbase + 32),
                        T_DIM, DK_DIM, 32, DK_DIM, T_DIM);
        }
#else
        {   // fallback: cooperative global->LDS copy (128 threads)
            const u32x4* kg = (const u32x4*)(kbase + (size_t)sbase * DK_DIM);
            u32x4* kl = (u32x4*)&ktile[buf][0];
#pragma unroll
            for (int i = 0; i < 4; ++i) kl[threadIdx.x + 128 * i] = kg[threadIdx.x + 128 * i];
            // V: 128 rows of 32 bf16 (64B) strided by T_DIM in global
            for (int row = threadIdx.x >> 2; row < DK_DIM; row += 32) {
                const u32x4* vg = (const u32x4*)(vbase + (size_t)row * T_DIM + sbase);
                ((u32x4*)&vtile[buf][row * 32])[threadIdx.x & 3] = vg[threadIdx.x & 3];
            }
            __syncthreads();
        }
#endif

        // --- S = Q @ K^T for 32 keys (two 16x16 C tiles), K frags from LDS ---
        v8f S[2];
#pragma unroll
        for (int jj = 0; jj < 2; ++jj) {
            v8f s = vzero;
            const unsigned short* krow = &ktile[buf][(jj * 16 + m) * DK_DIM + 8 * h];
#pragma unroll
            for (int kc = 0; kc < 4; ++kc) {
                BF16x16 kf;
                kf.q[0] = *(const u32x4*)(krow + kc * 32);
                kf.q[1] = *(const u32x4*)(krow + kc * 32 + 16);
                s = __builtin_amdgcn_wmma_f32_16x16x32_bf16(
                    false, qfr[kc], false, kf.v, (short)0, s, false, false);
            }
            S[jj] = s;
        }

        // --- faithful mask (before scale) + 1/sqrt(dk) scale ---
#pragma unroll
        for (int jj = 0; jj < 2; ++jj) {
            const int sg = sbase + jj * 16 + m;
#pragma unroll
            for (int r = 0; r < 8; ++r) {
                const int tg = tb * 16 + r + 8 * h;
                float v = S[jj][r];
                if (do_mask && sg <= tg) v += -1.0e9f;
                S[jj][r] = v * inv_sqrt;
            }
        }

        // --- online softmax (row stats in 16-lane halves) ---
        float mnew[8], corr[8], psum[8];
#pragma unroll
        for (int r = 0; r < 8; ++r) {
            float v = fmaxf(S[0][r], S[1][r]);
#pragma unroll
            for (int off = 1; off < 16; off <<= 1) v = fmaxf(v, __shfl_xor(v, off, 32));
            mnew[r] = fmaxf(mx[r], v);
            corr[r] = __expf(mx[r] - mnew[r]);
        }
#pragma unroll
        for (int r = 0; r < 8; ++r) {
            float p0 = __expf(S[0][r] - mnew[r]);
            float p1 = __expf(S[1][r] - mnew[r]);
            plds[wid][(r + 8 * h) * 32 + m]      = f2bf_rne(p0);
            plds[wid][(r + 8 * h) * 32 + 16 + m] = f2bf_rne(p1);
            float ps = p0 + p1;
#pragma unroll
            for (int off = 1; off < 16; off <<= 1) ps += __shfl_xor(ps, off, 32);
            psum[r] = ps;
        }
#pragma unroll
        for (int r = 0; r < 8; ++r) {
            ls[r] = ls[r] * corr[r] + psum[r];
            mx[r] = mnew[r];
        }

        // per-wave P tile: DS ops are in-order within a wave; make the
        // store->read ordering explicit (no block barrier needed)
        asm volatile("s_wait_dscnt 0" ::: "memory");

        BF16x16 pf;
        pf.q[0] = *(const u32x4*)(&plds[wid][m * 32 + 8 * h]);
        pf.q[1] = *(const u32x4*)(&plds[wid][m * 32 + 16 + 8 * h]);

        // --- O = O*corr + P @ V, V frags from LDS ---
#pragma unroll
        for (int nt = 0; nt < 8; ++nt) {
            v8f oo = o[nt];
#pragma unroll
            for (int r = 0; r < 8; ++r) oo[r] *= corr[r];
            const unsigned short* vrow = &vtile[buf][(nt * 16 + m) * 32 + 8 * h];
            BF16x16 vf;
            vf.q[0] = *(const u32x4*)(vrow);
            vf.q[1] = *(const u32x4*)(vrow + 16);
            o[nt] = __builtin_amdgcn_wmma_f32_16x16x32_bf16(
                false, pf.v, false, vf.v, (short)0, oo, false, false);
        }
#if !USE_TDM
        __syncthreads();
#endif
        // TDM path: next iteration's top barrier precedes the overwrite issue,
        // so no trailing barrier is required here.
    }

    // --- epilogue: normalize and write f32 output [b][t][128] ---
#pragma unroll
    for (int nt = 0; nt < 8; ++nt) {
#pragma unroll
        for (int r = 0; r < 8; ++r) {
            const int tg = tb * 16 + r + 8 * h;
            out[((size_t)b * T_DIM + tg) * DK_DIM + nt * 16 + m] = o[nt][r] / ls[r];
        }
    }
}

// ---------------------------------------------------------------------------
extern "C" void kernel_launch(void* const* d_in, const int* in_sizes, int n_in,
                              void* d_out, int out_size, void* d_ws, size_t ws_size,
                              hipStream_t stream) {
    (void)in_sizes; (void)n_in; (void)out_size; (void)ws_size;
    const float* in_q = (const float*)d_in[0];
    const float* in_k = (const float*)d_in[1];
    const float* in_v = (const float*)d_in[2];
    const float* Wq   = (const float*)d_in[3];
    const float* bq   = (const float*)d_in[4];
    const float* Wk   = (const float*)d_in[5];
    const float* bk   = (const float*)d_in[6];
    const float* Wv   = (const float*)d_in[7];
    const float* bv   = (const float*)d_in[8];
    const int*   mask = (const int*)d_in[9];
    float* out = (float*)d_out;

    // workspace: qbf 4MB | kbf 4MB | vT 4MB | WTq/WTk/WTv 3x128KB
    char* ws = (char*)d_ws;
    unsigned short* qbf = (unsigned short*)(ws);
    unsigned short* kbf = (unsigned short*)(ws + (size_t)4 * 1024 * 1024);
    unsigned short* vT  = (unsigned short*)(ws + (size_t)8 * 1024 * 1024);
    unsigned short* WTq = (unsigned short*)(ws + (size_t)12 * 1024 * 1024);
    unsigned short* WTk = WTq + (size_t)D_DIM * DK_DIM;
    unsigned short* WTv = WTk + (size_t)D_DIM * DK_DIM;

    wt_kernel<<<dim3(256, 3, 1), dim3(256), 0, stream>>>(Wq, Wk, Wv, WTq, WTk, WTv);
    proj_kernel<<<dim3(128, 8, 3), dim3(32), 0, stream>>>(in_q, in_k, in_v,
                                                          WTq, WTk, WTv,
                                                          bq, bk, bv,
                                                          qbf, kbf, vT);
    attn_kernel<<<dim3(32, 8), dim3(128), 0, stream>>>(qbf, kbf, vT, mask, out);
}
